// CachedAttention_79800492360203
// MI455X (gfx1250) — matrix operand order repaired and checked by reading, not compiled, not run on hardware
//
#include <hip/hip_runtime.h>
#include <hip/hip_bf16.h>
#include <math.h>

namespace {

constexpr int S_LEN  = 4096;
constexpr int DMODEL = 1024;
constexpr int NHEAD  = 16;
constexpr int DHEAD  = 64;
constexpr int EQKV   = 3072;

typedef unsigned short u16;
typedef __attribute__((ext_vector_type(16))) __bf16 v16bf;
typedef __attribute__((ext_vector_type(8)))  __bf16 v8bf;
typedef __attribute__((ext_vector_type(8)))  float  v8f;
typedef __attribute__((__vector_size__(16))) int v4i;

#if defined(__has_builtin)
#if __has_builtin(__builtin_amdgcn_global_load_async_to_lds_b128) && \
    __has_builtin(__builtin_amdgcn_s_wait_asynccnt)
#define USE_ASYNC_LDS 1
#endif
#endif

// 16-byte global -> LDS copy. Prefer the CDNA5 async path (no VGPR round trip,
// tracked by ASYNCcnt); fallback is a plain vector copy.
__device__ __forceinline__ void copy16_g2l(const u16* g, u16* l) {
#if USE_ASYNC_LDS
  __builtin_amdgcn_global_load_async_to_lds_b128(
      (__attribute__((address_space(1))) v4i*)g,
      (__attribute__((address_space(3))) v4i*)l, 0, 0);
#else
  *reinterpret_cast<uint4*>(l) = *reinterpret_cast<const uint4*>(g);
#endif
}

__device__ __forceinline__ void g2l_commit() {
#if USE_ASYNC_LDS
  __builtin_amdgcn_s_wait_asynccnt(0);   // s_wait_asynccnt 0
#endif
}

__device__ __forceinline__ u16 f2bf(float f) {
  union { float f; unsigned int u; } c; c.f = f;
  unsigned int u = c.u + 0x7FFFu + ((c.u >> 16) & 1u);   // round-to-nearest-even
  return (u16)(u >> 16);
}

__device__ __forceinline__ v8f wmma_bf16(v16bf a, v16bf b, v8f c) {
  // emits v_wmma_f32_16x16x32_bf16
  return __builtin_amdgcn_wmma_f32_16x16x32_bf16(false, a, false, b, (short)0, c,
                                                 false, false);
}

// A-matrix 16x32 bf16 fragment (ISA 7.12.2): lane L holds row m=L&15;
// K chunks [sel*8, sel*8+8) and [16+sel*8, 16+sel*8+8), sel=L>>4.
__device__ __forceinline__ v16bf load_a_frag(const u16* base, int ld,
                                             int row0, int k0, int lane) {
  const int m   = lane & 15;
  const int sel = lane >> 4;
  const u16* p = base + (row0 + m) * ld + k0 + sel * 8;
  v8bf lo = *reinterpret_cast<const v8bf*>(p);
  v8bf hi = *reinterpret_cast<const v8bf*>(p + 16);
  v16bf a;
#pragma unroll
  for (int i = 0; i < 8; ++i) { a[i] = lo[i]; a[i + 8] = hi[i]; }
  return a;
}

// B-matrix 32x16 bf16 fragment: lane L holds column n=L&15;
// K chunks [sel*8, sel*8+8) and [16+sel*8, 16+sel*8+8), sel=L>>4.
__device__ __forceinline__ v16bf load_b_frag(const u16* base, int ld,
                                             int k0, int n0, int lane) {
  const int n   = lane & 15;
  const int sel = lane >> 4;
  const __bf16* p = reinterpret_cast<const __bf16*>(base) + (size_t)(k0 + sel * 8) * ld + n0 + n;
  v16bf b;
#pragma unroll
  for (int i = 0; i < 8; ++i) { b[i] = p[(size_t)i * ld]; b[i + 8] = p[(size_t)(16 + i) * ld]; }
  return b;
}

// ---------------- conversion kernels ----------------

__global__ void k_cvt(const float* __restrict__ in, u16* __restrict__ out, int n) {
  int i = blockIdx.x * blockDim.x + threadIdx.x;
  if (i < n) out[i] = f2bf(in[i]);
}

// in [rows][cols] -> out [cols][rows]  (bf16, K-major weights for the B operand)
__global__ void k_cvt_transpose(const float* __restrict__ in, u16* __restrict__ out,
                                int rows, int cols) {
  int i = blockIdx.x * blockDim.x + threadIdx.x;
  if (i < rows * cols) {
    int r = i / cols, c = i % cols;
    out[c * rows + r] = f2bf(in[i]);
  }
}

// ---------------- QKV projection GEMM ----------------
// C[s,e] = sum_d x[s,d]*Wqkv[e,d].  128x128 block tile, K-stage 64, 8 waves in
// 2x4 grid; each wave owns a 64x32 sub-block = 4x2 WMMA tiles -> 16 WMMA/stage.
// Q,V written [h][s][dh]; K written TRANSPOSED [h][dh][s] so the attention
// B-operand (lane = key column, elements = dh rows; 16 lanes read contiguous keys) loads directly from global.
__global__ __launch_bounds__(256)
void k_qkv_gemm(const u16* __restrict__ Xb,    // [S][1024]
                const u16* __restrict__ Wt,    // [1024][3072]
                u16* __restrict__ QKV) {       // Q[16][S][64], Kt[16][64][S], V[16][S][64]
  __shared__ u16 As[128 * 64];
  __shared__ u16 Bs[64 * 128];
  const int bm = blockIdx.x, bn = blockIdx.y;
  const int tid = threadIdx.x, w = tid >> 5, lane = tid & 31;
  const int wm = w >> 2, wn = w & 3;
  const int hi = lane >> 4, col = lane & 15;
  const int arow = tid >> 1, acb = (tid & 1) * 32;
  const int brow = tid >> 2, bcb = (tid & 3) * 32;
  v8f acc[4][2];
#pragma unroll
  for (int mt = 0; mt < 4; ++mt)
#pragma unroll
    for (int nt = 0; nt < 2; ++nt)
#pragma unroll
      for (int i = 0; i < 8; ++i) acc[mt][nt][i] = 0.f;

  for (int d0 = 0; d0 < DMODEL; d0 += 64) {
    __syncthreads();
#pragma unroll
    for (int j = 0; j < 4; ++j)
      copy16_g2l(&Xb[(size_t)(bm * 128 + arow) * DMODEL + d0 + acb + j * 8],
                 &As[arow * 64 + acb + j * 8]);
#pragma unroll
    for (int j = 0; j < 4; ++j)
      copy16_g2l(&Wt[(size_t)(d0 + brow) * EQKV + bn * 128 + bcb + j * 8],
                 &Bs[brow * 128 + bcb + j * 8]);
    if (d0 + 64 < DMODEL) {
      __builtin_prefetch(&Xb[(size_t)(bm * 128 + arow) * DMODEL + d0 + 64 + acb], 0, 3);
      __builtin_prefetch(&Wt[(size_t)(d0 + 64 + brow) * EQKV + bn * 128 + bcb], 0, 3);
    }
    g2l_commit();
    __syncthreads();
#pragma unroll
    for (int ks = 0; ks < 64; ks += 32) {
      v16bf af[4];
#pragma unroll
      for (int mt = 0; mt < 4; ++mt)
        af[mt] = load_a_frag(As, 64, (wm * 4 + mt) * 16, ks, lane);
      v16bf b0 = load_b_frag(Bs, 128, ks, (wn * 2 + 0) * 16, lane);
      v16bf b1 = load_b_frag(Bs, 128, ks, (wn * 2 + 1) * 16, lane);
#pragma unroll
      for (int mt = 0; mt < 4; ++mt) {
        acc[mt][0] = wmma_bf16(af[mt], b0, acc[mt][0]);
        acc[mt][1] = wmma_bf16(af[mt], b1, acc[mt][1]);
      }
    }
  }
  // epilogue: scatter into Q/Kt/V sections
#pragma unroll
  for (int mt = 0; mt < 4; ++mt)
#pragma unroll
    for (int nt = 0; nt < 2; ++nt)
#pragma unroll
      for (int i = 0; i < 8; ++i) {
        float v = acc[mt][nt][i];
        int srow = bm * 128 + (wm * 4 + mt) * 16 + i + 8 * hi;
        int e = bn * 128 + (wn * 2 + nt) * 16 + col;
        int sec = e >> 10, r = e & 1023, h = r >> 6, dh = r & 63;
        size_t off;
        if (sec == 1)   // K transposed: [h][dh][s]
          off = (size_t)NHEAD * S_LEN * DHEAD + ((size_t)(h * DHEAD + dh)) * S_LEN + srow;
        else            // Q (sec 0) / V (sec 2): [h][s][dh]
          off = ((size_t)(sec * NHEAD + h) * S_LEN + srow) * DHEAD + dh;
        QKV[off] = f2bf(v);
      }
}

// ---------------- flash attention ----------------
// grid (S/64, H), 128 threads = 4 waves; wave w owns query rows [w*16, w*16+16).
// No inner barriers: K^T and V B-fragments come straight from global; the P
// staging buffer is wave-private (per-wave LDS ops are in-order, ISA 7.3).
__global__ __launch_bounds__(128)
void k_attn(const u16* __restrict__ QKV,
            u16* __restrict__ AOut) {         // [S][1024] bf16
  __shared__ u16 Qs[64 * 64];                 // [q][dh]
  __shared__ u16 Ps[64 * 64];                 // [q][key], rows w*16.. are wave-private
  const int qb = blockIdx.x * 64;
  const int h  = blockIdx.y;
  const int tid = threadIdx.x, w = tid >> 5, lane = tid & 31;
  const int hi = lane >> 4, col = lane & 15;
  const u16* Qg  = QKV + (size_t)h * S_LEN * DHEAD;
  const u16* Ktg = QKV + (size_t)NHEAD * S_LEN * DHEAD + (size_t)h * DHEAD * S_LEN; // [dh][s]
  const u16* Vg  = QKV + (size_t)2 * NHEAD * S_LEN * DHEAD + (size_t)h * S_LEN * DHEAD;

  {  // stage Q block via async global->LDS
    int row = tid >> 1, cb = (tid & 1) * 32;
#pragma unroll
    for (int j = 0; j < 4; ++j)
      copy16_g2l(&Qg[(size_t)(qb + row) * DHEAD + cb + j * 8],
                 &Qs[row * 64 + cb + j * 8]);
  }
  g2l_commit();
  __syncthreads();

  v8f acc[4]; float mrow[8], lrow[8];
#pragma unroll
  for (int t = 0; t < 4; ++t)
#pragma unroll
    for (int i = 0; i < 8; ++i) acc[t][i] = 0.f;
#pragma unroll
  for (int i = 0; i < 8; ++i) { mrow[i] = -INFINITY; lrow[i] = 0.f; }

  const int nkb = blockIdx.x + 1;   // causal: key blocks 0..qb
  for (int kbi = 0; kbi < nkb; ++kbi) {
    const int kb = kbi * 64;
    if (kbi + 1 < nkb) {            // prefetch next key block
      __builtin_prefetch(&Ktg[(size_t)lane * S_LEN + kb + 64], 0, 3);
      __builtin_prefetch(&Ktg[(size_t)(32 + lane) * S_LEN + kb + 64], 0, 3);
      __builtin_prefetch(&Vg[(size_t)(kb + 64 + lane) * DHEAD], 0, 3);
      __builtin_prefetch(&Vg[(size_t)(kb + 96 + lane) * DHEAD], 0, 3);
    }
    // S = Q K^T  (dh=64 -> two K=32 WMMA steps per 16x16 tile)
    v8f s[4];
#pragma unroll
    for (int nt = 0; nt < 4; ++nt) {
      v8f c;
#pragma unroll
      for (int i = 0; i < 8; ++i) c[i] = 0.f;
#pragma unroll
      for (int ks = 0; ks < 64; ks += 32) {
        v16bf a = load_a_frag(Qs, 64, w * 16, ks, lane);
        v16bf b = load_b_frag(Ktg, S_LEN, ks, kb + nt * 16, lane);  // global K^T
        c = wmma_bf16(a, b, c);
      }
      s[nt] = c;
    }
    // scale + causal mask (C layout: row = i+8*hi, col = lane&15)
#pragma unroll
    for (int nt = 0; nt < 4; ++nt)
#pragma unroll
      for (int i = 0; i < 8; ++i) {
        int q = qb + w * 16 + i + 8 * hi;
        int k = kb + nt * 16 + col;
        float v = s[nt][i] * 0.125f;
        s[nt][i] = (k > q) ? -INFINITY : v;
      }
    // online softmax: row reductions across the 16-lane column group
#pragma unroll
    for (int i = 0; i < 8; ++i) {
      float m = fmaxf(fmaxf(s[0][i], s[1][i]), fmaxf(s[2][i], s[3][i]));
#pragma unroll
      for (int off = 1; off < 16; off <<= 1)
        m = fmaxf(m, __shfl_xor(m, off, 32));
      float newm = fmaxf(mrow[i], m);
      float sc = __expf(mrow[i] - newm);
      float rs = 0.f;
#pragma unroll
      for (int nt = 0; nt < 4; ++nt) {
        float p = __expf(s[nt][i] - newm);
        rs += p;
        Ps[(w * 16 + i + 8 * hi) * 64 + nt * 16 + col] = f2bf(p);
      }
#pragma unroll
      for (int off = 1; off < 16; off <<= 1)
        rs += __shfl_xor(rs, off, 32);
      lrow[i] = lrow[i] * sc + rs;
      mrow[i] = newm;
#pragma unroll
      for (int nt = 0; nt < 4; ++nt) acc[nt][i] *= sc;
    }
    // O += P V : B operand is V in natural [key][dh] layout -> global loads
#pragma unroll
    for (int nt = 0; nt < 4; ++nt)
#pragma unroll
      for (int ks = 0; ks < 64; ks += 32) {
        v16bf a = load_a_frag(Ps, 64, w * 16, ks, lane);
        v16bf b = load_b_frag(Vg, DHEAD, kb + ks, nt * 16, lane);
        acc[nt] = wmma_bf16(a, b, acc[nt]);
      }
  }
  // epilogue: normalize, write bf16 attn output in [s][h*64+dh] layout
#pragma unroll
  for (int nt = 0; nt < 4; ++nt)
#pragma unroll
    for (int i = 0; i < 8; ++i) {
      int q = qb + w * 16 + i + 8 * hi;
      int dcol = h * DHEAD + nt * 16 + col;
      AOut[(size_t)q * DMODEL + dcol] = f2bf(acc[nt][i] / lrow[i]);
    }
}

// ---------------- output projection GEMM ----------------
__global__ __launch_bounds__(256)
void k_out_gemm(const u16* __restrict__ Ab,   // [S][1024] bf16
                const u16* __restrict__ Wt,   // [1024][1024] bf16 (d-major)
                float* __restrict__ Out) {    // [S][1024] f32
  __shared__ u16 As[128 * 64];
  __shared__ u16 Bs[64 * 128];
  const int bm = blockIdx.x, bn = blockIdx.y;
  const int tid = threadIdx.x, w = tid >> 5, lane = tid & 31;
  const int wm = w >> 2, wn = w & 3;
  const int hi = lane >> 4, col = lane & 15;
  const int arow = tid >> 1, acb = (tid & 1) * 32;
  const int brow = tid >> 2, bcb = (tid & 3) * 32;
  v8f acc[4][2];
#pragma unroll
  for (int mt = 0; mt < 4; ++mt)
#pragma unroll
    for (int nt = 0; nt < 2; ++nt)
#pragma unroll
      for (int i = 0; i < 8; ++i) acc[mt][nt][i] = 0.f;

  for (int d0 = 0; d0 < DMODEL; d0 += 64) {
    __syncthreads();
#pragma unroll
    for (int j = 0; j < 4; ++j)
      copy16_g2l(&Ab[(size_t)(bm * 128 + arow) * DMODEL + d0 + acb + j * 8],
                 &As[arow * 64 + acb + j * 8]);
#pragma unroll
    for (int j = 0; j < 4; ++j)
      copy16_g2l(&Wt[(size_t)(d0 + brow) * DMODEL + bn * 128 + bcb + j * 8],
                 &Bs[brow * 128 + bcb + j * 8]);
    if (d0 + 64 < DMODEL) {
      __builtin_prefetch(&Ab[(size_t)(bm * 128 + arow) * DMODEL + d0 + 64 + acb], 0, 3);
      __builtin_prefetch(&Wt[(size_t)(d0 + 64 + brow) * DMODEL + bn * 128 + bcb], 0, 3);
    }
    g2l_commit();
    __syncthreads();
#pragma unroll
    for (int ks = 0; ks < 64; ks += 32) {
      v16bf af[4];
#pragma unroll
      for (int mt = 0; mt < 4; ++mt)
        af[mt] = load_a_frag(As, 64, (wm * 4 + mt) * 16, ks, lane);
      v16bf b0 = load_b_frag(Bs, 128, ks, (wn * 2 + 0) * 16, lane);
      v16bf b1 = load_b_frag(Bs, 128, ks, (wn * 2 + 1) * 16, lane);
#pragma unroll
      for (int mt = 0; mt < 4; ++mt) {
        acc[mt][0] = wmma_bf16(af[mt], b0, acc[mt][0]);
        acc[mt][1] = wmma_bf16(af[mt], b1, acc[mt][1]);
      }
    }
  }
#pragma unroll
  for (int mt = 0; mt < 4; ++mt)
#pragma unroll
    for (int nt = 0; nt < 2; ++nt)
#pragma unroll
      for (int i = 0; i < 8; ++i) {
        int srow = bm * 128 + (wm * 4 + mt) * 16 + i + 8 * hi;
        int e = bn * 128 + (wn * 2 + nt) * 16 + col;
        Out[(size_t)srow * DMODEL + e] = acc[mt][nt][i];
      }
}

}  // namespace

extern "C" void kernel_launch(void* const* d_in, const int* in_sizes, int n_in,
                              void* d_out, int out_size, void* d_ws, size_t ws_size,
                              hipStream_t stream) {
  (void)in_sizes; (void)n_in; (void)out_size; (void)ws_size;
  const float* x    = (const float*)d_in[0];
  const float* wqkv = (const float*)d_in[1];
  const float* wo   = (const float*)d_in[2];
  float* out = (float*)d_out;

  char* ws = (char*)d_ws;
  u16* Xb   = (u16*)(ws);                               // 8 MB
  u16* Wqt  = (u16*)(ws + (size_t)8  * 1024 * 1024);    // 6 MB
  u16* Wot  = (u16*)(ws + (size_t)14 * 1024 * 1024);    // 2 MB
  u16* QKV  = (u16*)(ws + (size_t)16 * 1024 * 1024);    // 24 MB (Q, K^T, V)
  u16* AOut = (u16*)(ws + (size_t)40 * 1024 * 1024);    // 8 MB

  k_cvt<<<(S_LEN * DMODEL) / 256, 256, 0, stream>>>(x, Xb, S_LEN * DMODEL);
  k_cvt_transpose<<<(EQKV * DMODEL) / 256, 256, 0, stream>>>(wqkv, Wqt, EQKV, DMODEL);
  k_cvt_transpose<<<(DMODEL * DMODEL) / 256, 256, 0, stream>>>(wo, Wot, DMODEL, DMODEL);

  k_qkv_gemm<<<dim3(S_LEN / 128, EQKV / 128), 256, 0, stream>>>(Xb, Wqt, QKV);
  k_attn<<<dim3(S_LEN / 64, NHEAD), 128, 0, stream>>>(QKV, AOut);
  k_out_gemm<<<dim3(S_LEN / 128, DMODEL / 128), 256, 0, stream>>>(AOut, Wot, out);
}
